// LocalFeatureTransformer_48533130444976
// MI455X (gfx1250) — compile-verified, matrix operand order checked by reading, not run on hardware
//
#include <hip/hip_runtime.h>

// ---------------------------------------------------------------------------
// LoFTR-style LocalFeatureTransformer for MI455X (gfx1250, wave32, WMMA).
// GEMMs run on v_wmma_f32_16x16x32_bf16 (bf16 in / f32 accumulate),
// register-staged double-buffered LDS pipeline, one barrier per K-step.
// ---------------------------------------------------------------------------

#define NBATCH 4
#define SEQ    4800
#define CDIM   256
#define NHEADS 8
#define HD     32
#define MROWS  (NBATCH * SEQ)            // 19200 rows (divisible by 128)
#define BLC    ((size_t)MROWS * CDIM)    // 4,915,200 floats per feature map

typedef __attribute__((ext_vector_type(16))) __bf16 bf16x16;
typedef __attribute__((ext_vector_type(8)))  float  f32x8;

// ---------------------------------------------------------------------------
// GEMM: Y = act(A @ W).  A: MxK f32 row-major, W: KxN f32 row-major.
// Block tile 128x128, BK=32, 256 threads (8 wave32s), wave tile 32x64
// (2x4 v_wmma_f32_16x16x32_bf16 accumulators).  Double-buffered LDS.
// MODE: 0 = identity, 1 = relu, 2 = elu(x)+1  (linear-attention phi)
// ---------------------------------------------------------------------------
template <int MODE>
__global__ __launch_bounds__(256) void gemm_bf16_wmma(
    const float* __restrict__ A, const float* __restrict__ W,
    float* __restrict__ Y, int M, int K, int N)
{
  __shared__ __bf16 As[2][128][32];   // row-major A tiles (ping-pong)
  __shared__ __bf16 Bs[2][128][32];   // W tiles stored transposed: Bs[.][n][k]

  const int tid  = threadIdx.x;
  const int lane = tid & 31;
  const int wave = tid >> 5;
  const int m0 = blockIdx.x * 128;
  const int n0 = blockIdx.y * 128;
  const int wm = (wave & 3) * 32;      // wave row offset within block tile
  const int wn = (wave >> 2) * 64;     // wave col offset within block tile
  const int lrow = lane & 15;
  const int kgrp = (lane >> 4) << 3;   // K-group base: 0 (lanes 0-15) / 8 (16-31)

  const int lr = tid >> 1;             // 0..127: row (A) / col (W) per loader thread
  const int lc = (tid & 1) << 4;       // 0 or 16: k-chunk per loader thread

  f32x8 acc[2][4];
#pragma unroll
  for (int i = 0; i < 2; ++i)
#pragma unroll
    for (int j = 0; j < 4; ++j)
      acc[i][j] = f32x8{};

  float4 areg[4];
  float  breg[16];

  // --- global -> register staging of one 128x32 A slab + 32x128 W slab ---
  auto load_tile = [&](int kb) {
    const float* ap = A + (size_t)(m0 + lr) * K + kb + lc;
#pragma unroll
    for (int q = 0; q < 4; ++q)
      areg[q] = *(const float4*)(ap + 4 * q);
    const float* wp = W + (size_t)(kb + lc) * N + (n0 + lr);
#pragma unroll
    for (int j = 0; j < 16; ++j)
      breg[j] = wp[(size_t)j * N];     // coalesced across threads (lr contiguous)
  };
  // --- register -> LDS (with f32 -> bf16 convert) ---
  auto store_tile = [&](int buf) {
#pragma unroll
    for (int q = 0; q < 4; ++q) {
      As[buf][lr][lc + 4 * q + 0] = (__bf16)areg[q].x;
      As[buf][lr][lc + 4 * q + 1] = (__bf16)areg[q].y;
      As[buf][lr][lc + 4 * q + 2] = (__bf16)areg[q].z;
      As[buf][lr][lc + 4 * q + 3] = (__bf16)areg[q].w;
    }
#pragma unroll
    for (int j = 0; j < 16; ++j)
      Bs[buf][lr][lc + j] = (__bf16)breg[j];
  };

  load_tile(0);
  store_tile(0);
  __syncthreads();

  int buf = 0;
  for (int kb = 0; kb < K; kb += 32) {
    const bool last = (kb + 32 >= K);
    if (!last) load_tile(kb + 32);     // next slab's global loads in flight

    // --- build fragments per ISA 16-bit A/B layouts ---
    // lane half selects K groups {0-7,16-23} vs {8-15,24-31}; element e ->
    // k = kgrp + (e>>3)*16 + (e&7): contiguous runs of 8 -> ds_load_b128 pairs.
    bf16x16 afrag[2];
#pragma unroll
    for (int i = 0; i < 2; ++i) {
      const __bf16* arow = &As[buf][wm + i * 16 + lrow][0];
#pragma unroll
      for (int e = 0; e < 16; ++e)
        afrag[i][e] = arow[kgrp + ((e >> 3) << 4) + (e & 7)];
    }
    bf16x16 bfrag[4];
#pragma unroll
    for (int j = 0; j < 4; ++j) {
      const __bf16* brow = &Bs[buf][wn + j * 16 + lrow][0];
#pragma unroll
      for (int e = 0; e < 16; ++e)
        bfrag[j][e] = brow[kgrp + ((e >> 3) << 4) + (e & 7)];
    }

#pragma unroll
    for (int i = 0; i < 2; ++i)
#pragma unroll
      for (int j = 0; j < 4; ++j)
        acc[i][j] = __builtin_amdgcn_wmma_f32_16x16x32_bf16(
            false, afrag[i], false, bfrag[j], (short)0, acc[i][j], false, false);

    if (!last) store_tile(buf ^ 1);    // fill other buffer; current stays readable
    __syncthreads();                   // one barrier per K-step
    buf ^= 1;
  }

  // --- epilogue (branch-free per MODE): VGPR r -> M = r (+8 lanes 16-31) ---
  const int rsel = (lane >> 4) << 3;
  const int csel = lane & 15;
#pragma unroll
  for (int i = 0; i < 2; ++i)
#pragma unroll
    for (int j = 0; j < 4; ++j) {
      float* yp = Y + (size_t)(m0 + wm + i * 16 + rsel) * N
                    + (n0 + wn + j * 16 + csel);
#pragma unroll
      for (int r = 0; r < 8; ++r) {
        float v = acc[i][j][r];
        if (MODE == 1)      v = fmaxf(v, 0.0f);
        else if (MODE == 2) v = (v > 0.0f) ? (v + 1.0f) : __expf(v);  // elu+1
        yp[(size_t)r * N] = v;
      }
    }
}

// ---------------------------------------------------------------------------
// KV[b,h,d,v] = (1/S) * sum_s K[b,s,h,d] * V[b,s,h,v];  Ksum[b,h,d] = sum_s K
// One 1024-thread block per (b,h); 32-row LDS tiles.
// ---------------------------------------------------------------------------
__global__ __launch_bounds__(1024) void kv_reduce(
    const float* __restrict__ Kp, const float* __restrict__ V,
    float* __restrict__ KV, float* __restrict__ Ksum)
{
  const int bh = blockIdx.x;
  const int b  = bh >> 3;
  const int h  = bh & 7;
  const int tid = threadIdx.x;
  const int d = tid >> 5;
  const int v = tid & 31;
  __shared__ float Ks[32][33];
  __shared__ float Vs[32][33];
  float acc = 0.0f, ks = 0.0f;
  for (int s0 = 0; s0 < SEQ; s0 += 32) {
    size_t g = ((size_t)b * SEQ + (s0 + d)) * CDIM + h * HD + v;
    Ks[d][v] = Kp[g];
    Vs[d][v] = V[g];
    __syncthreads();
#pragma unroll
    for (int s = 0; s < 32; ++s)
      acc += Ks[s][d] * Vs[s][v];
    if (v == 0) {
#pragma unroll
      for (int s = 0; s < 32; ++s) ks += Ks[s][d];
    }
    __syncthreads();
  }
  KV[((size_t)bh * HD + d) * HD + v] = acc * (1.0f / (float)SEQ);
  if (v == 0) Ksum[(size_t)bh * HD + d] = ks;
}

// ---------------------------------------------------------------------------
// msg[b,l,h,v] = (Q . KV[:,v]) * S / (Q . Ksum + eps).  One block per token.
// ---------------------------------------------------------------------------
__global__ __launch_bounds__(256) void attn_apply(
    const float* __restrict__ Qp, const float* __restrict__ KV,
    const float* __restrict__ Ksum, float* __restrict__ Msg)
{
  const int row = blockIdx.x;            // b*SEQ + l
  const int b = row / SEQ;
  const int tid = threadIdx.x;
  const int h = tid >> 5;
  const int v = tid & 31;
  __shared__ float Qs[CDIM];
  Qs[tid] = Qp[(size_t)row * CDIM + tid];
  __syncthreads();
  const float* kvh = KV + (size_t)(b * NHEADS + h) * HD * HD;  // L2-resident
  const float* ksh = Ksum + (size_t)(b * NHEADS + h) * HD;
  float z = 0.0f, m = 0.0f;
#pragma unroll
  for (int dd = 0; dd < HD; ++dd) {
    float q = Qs[h * HD + dd];
    z += q * ksh[dd];
    m += q * kvh[dd * HD + v];
  }
  Msg[(size_t)row * CDIM + tid] = m * ((float)SEQ / (z + 1e-6f));
}

// ---------------------------------------------------------------------------
// LayerNorm over C=256 (one block per row); optional residual add after LN.
// ---------------------------------------------------------------------------
__global__ __launch_bounds__(256) void ln_kernel(
    const float* __restrict__ X, const float* __restrict__ gw,
    const float* __restrict__ gb, const float* __restrict__ Res,
    float* __restrict__ Y)
{
  const int row = blockIdx.x;
  const int tid = threadIdx.x;
  __shared__ float red[256];
  float x = X[(size_t)row * CDIM + tid];
  red[tid] = x;
  __syncthreads();
#pragma unroll
  for (int s = 128; s > 0; s >>= 1) {
    if (tid < s) red[tid] += red[tid + s];
    __syncthreads();
  }
  float mu = red[0] * (1.0f / CDIM);
  __syncthreads();
  float dx = x - mu;
  red[tid] = dx * dx;
  __syncthreads();
#pragma unroll
  for (int s = 128; s > 0; s >>= 1) {
    if (tid < s) red[tid] += red[tid + s];
    __syncthreads();
  }
  float var = red[0] * (1.0f / CDIM);
  float y = dx * rsqrtf(var + 1e-5f) * gw[tid] + gb[tid];
  if (Res) y += Res[(size_t)row * CDIM + tid];
  Y[(size_t)row * CDIM + tid] = y;
}

// ---------------------------------------------------------------------------
// H[m, 0:256] = X[m,:], H[m, 256:512] = Msg[m,:]
// ---------------------------------------------------------------------------
__global__ __launch_bounds__(256) void concat_kernel(
    const float* __restrict__ X, const float* __restrict__ Msg,
    float* __restrict__ Hc)
{
  size_t idx = (size_t)blockIdx.x * 256 + threadIdx.x;
  size_t row = idx >> 9;            // / 512
  int col = (int)(idx & 511);
  Hc[idx] = (col < CDIM) ? X[row * CDIM + col]
                         : Msg[row * CDIM + (col - CDIM)];
}

// ---------------------------------------------------------------------------
// Host-side orchestration
// ---------------------------------------------------------------------------
struct LayerW {
  const float *Wq, *Wk, *Wv, *Wm, *W1, *W2, *l1w, *l1b, *l2w, *l2b;
};

static inline void run_gemm(const float* A, const float* W, float* Y,
                            int M, int K, int N, int mode, hipStream_t s) {
  dim3 grid(M / 128, N / 128);
  if (mode == 0)
    gemm_bf16_wmma<0><<<grid, 256, 0, s>>>(A, W, Y, M, K, N);
  else if (mode == 1)
    gemm_bf16_wmma<1><<<grid, 256, 0, s>>>(A, W, Y, M, K, N);
  else
    gemm_bf16_wmma<2><<<grid, 256, 0, s>>>(A, W, Y, M, K, N);
}

// Workspace layout (floats):
//   [0, BLC)        tQ   -> reused as tH[0:BLC)
//   [BLC, 2BLC)     tK   -> reused as tH[BLC:2BLC)
//   [2BLC, 3BLC)    tV   -> reused as tH1[0:BLC)
//   [3BLC, 4BLC)    tMsg -> reused as tH1[BLC:2BLC)
//   [4BLC, 5BLC)    tMsg2 -> reused as tH2
//   [5BLC, ...)     KV (B*H*32*32) + Ksum (B*H*32)
// Total ~ 5*BLC + 33K floats ~ 99 MB.
static void encoder_layer(float* x, const float* src, const LayerW& w,
                          float* ws, hipStream_t s) {
  float* tQ   = ws;
  float* tK   = ws + BLC;
  float* tV   = ws + 2 * BLC;
  float* tMsg = ws + 3 * BLC;
  float* tM2  = ws + 4 * BLC;
  float* tH   = ws;            // 2*BLC, overwrites tQ/tK (dead)
  float* tH1  = ws + 2 * BLC;  // 2*BLC, overwrites tV/tMsg (dead)
  float* tH2  = ws + 4 * BLC;  // overwrites tM2 (dead)
  float* kv   = ws + 5 * BLC;
  float* ksum = kv + (size_t)NBATCH * NHEADS * HD * HD;

  run_gemm(x,   w.Wq, tQ, MROWS, CDIM, CDIM, 2, s);   // phi(x @ Wq)
  run_gemm(src, w.Wk, tK, MROWS, CDIM, CDIM, 2, s);   // phi(src @ Wk)
  run_gemm(src, w.Wv, tV, MROWS, CDIM, CDIM, 0, s);   // src @ Wv
  kv_reduce<<<NBATCH * NHEADS, 1024, 0, s>>>(tK, tV, kv, ksum);
  attn_apply<<<MROWS, 256, 0, s>>>(tQ, kv, ksum, tMsg);
  run_gemm(tMsg, w.Wm, tM2, MROWS, CDIM, CDIM, 0, s); // msg @ Wm
  ln_kernel<<<MROWS, 256, 0, s>>>(tM2, w.l1w, w.l1b, nullptr, tMsg);
  concat_kernel<<<(int)((size_t)MROWS * 512 / 256), 256, 0, s>>>(x, tMsg, tH);
  run_gemm(tH,  w.W1, tH1, MROWS, 2 * CDIM, 2 * CDIM, 1, s);  // relu(h @ W1)
  run_gemm(tH1, w.W2, tH2, MROWS, 2 * CDIM, CDIM, 0, s);      // @ W2
  ln_kernel<<<MROWS, 256, 0, s>>>(tH2, w.l2w, w.l2b, /*res=*/x, /*out=*/x);
}

extern "C" void kernel_launch(void* const* d_in, const int* in_sizes, int n_in,
                              void* d_out, int out_size, void* d_ws, size_t ws_size,
                              hipStream_t stream) {
  const float* feat0 = (const float*)d_in[0];
  const float* feat1 = (const float*)d_in[1];
  const float* Wq = (const float*)d_in[2];
  const float* Wk = (const float*)d_in[3];
  const float* Wv = (const float*)d_in[4];
  const float* Wm = (const float*)d_in[5];
  const float* W1 = (const float*)d_in[6];
  const float* W2 = (const float*)d_in[7];
  const float* l1w = (const float*)d_in[8];
  const float* l1b = (const float*)d_in[9];
  const float* l2w = (const float*)d_in[10];
  const float* l2b = (const float*)d_in[11];

  // d_out doubles as the ping-pong feature storage: [F0 | F1]
  float* F0 = (float*)d_out;
  float* F1 = F0 + BLC;
  hipMemcpyAsync(F0, feat0, BLC * sizeof(float), hipMemcpyDeviceToDevice, stream);
  hipMemcpyAsync(F1, feat1, BLC * sizeof(float), hipMemcpyDeviceToDevice, stream);

  float* ws = (float*)d_ws;
  const size_t cc = (size_t)CDIM * CDIM;

  for (int i = 0; i < 8; ++i) {
    LayerW w = {
      Wq + (size_t)i * cc, Wk + (size_t)i * cc,
      Wv + (size_t)i * cc, Wm + (size_t)i * cc,
      W1 + (size_t)i * 4 * cc,      // (2C)x(2C)
      W2 + (size_t)i * 2 * cc,      // (2C)x(C)
      l1w + (size_t)i * CDIM, l1b + (size_t)i * CDIM,
      l2w + (size_t)i * CDIM, l2b + (size_t)i * CDIM,
    };
    if ((i & 1) == 0) {            // 'self'
      encoder_layer(F0, F0, w, ws, stream);
      encoder_layer(F1, F1, w, ws, stream);
    } else {                       // 'cross' (sequential, uses updated F0)
      encoder_layer(F0, F1, w, ws, stream);
      encoder_layer(F1, F0, w, ws, stream);
    }
  }
}
